// Attention_8718783611482
// MI455X (gfx1250) — matrix operand order repaired and checked by reading, not compiled, not run on hardware
//
#include <hip/hip_runtime.h>

typedef __attribute__((ext_vector_type(16))) _Float16 v16h;
typedef __attribute__((ext_vector_type(8)))  _Float16 v8h;
typedef __attribute__((ext_vector_type(8)))  float    v8f;
typedef __attribute__((ext_vector_type(4)))  float    v4f;
typedef __attribute__((ext_vector_type(4)))  _Float16 v4h;

#define NQ    8192
#define MK    8192
#define DH    64
#define BM    64     // query rows per block (4 waves x 16)
#define BN    64     // key rows per tile
#define NWAVE 4

// DPP16 ROW_XMASK lane-xor move within 16-lane rows (wave32) — pure VALU shuffle.
template <int CTRL>
__device__ __forceinline__ float dpp_mov(float v) {
  int r = __builtin_amdgcn_update_dpp(0, __builtin_bit_cast(int, v),
                                      CTRL, 0xF, 0xF, true);
  return __builtin_bit_cast(float, r);
}
__device__ __forceinline__ float redmax16(float v) {
  v = fmaxf(v, dpp_mov<0x161>(v));  // xor 1
  v = fmaxf(v, dpp_mov<0x162>(v));  // xor 2
  v = fmaxf(v, dpp_mov<0x164>(v));  // xor 4
  v = fmaxf(v, dpp_mov<0x168>(v));  // xor 8
  return v;
}
__device__ __forceinline__ float redsum16(float v) {
  v += dpp_mov<0x161>(v);
  v += dpp_mov<0x162>(v);
  v += dpp_mov<0x164>(v);
  v += dpp_mov<0x168>(v);
  return v;
}

__global__ __launch_bounds__(128) void fa_wmma_kernel(
    const float* __restrict__ Qg, const float* __restrict__ Kg,
    const float* __restrict__ Vg, float* __restrict__ Og)
{
  __shared__ _Float16 sK [BN * DH];          // [key][d]   row-major, f16
  __shared__ _Float16 sVT[DH * BN];          // [dv][key]  TRANSPOSED, f16
  __shared__ _Float16 sP [NWAVE][16 * BN];   // per-wave P scratch [qrow][key]

  const int tid  = threadIdx.x;
  const int wave = tid >> 5;
  const int lane = tid & 31;
  const int ln   = lane & 15;   // N index within 16-wide tile
  const int hf   = lane >> 4;   // half-wave select
  const int qRow0 = blockIdx.x * BM + wave * 16;

  // ---- Q A-fragments: two 16x32 f16 fragments; each is two contiguous 8-elt runs ----
  // amap: k = e + 16*hf (contiguous)
  v16h qa[2];
  {
    const float* qrow = Qg + (qRow0 + ln) * DH;
#pragma unroll
    for (int c = 0; c < 2; ++c) {
      const int b0 = 32 * c + 16 * hf;
      const int b1 = 32 * c + 16 * hf + 8;
#pragma unroll
      for (int j = 0; j < 8; ++j) qa[c][j]     = (_Float16)qrow[b0 + j];
#pragma unroll
      for (int j = 0; j < 8; ++j) qa[c][8 + j] = (_Float16)qrow[b1 + j];
    }
  }

  v8f o[4];
#pragma unroll
  for (int n = 0; n < 4; ++n) o[n] = {};
  float mrow[8], lrow[8];     // running max (base-2 domain) and running sum
#pragma unroll
  for (int r = 0; r < 8; ++r) { mrow[r] = -__builtin_inff(); lrow[r] = 0.0f; }

  // softmax logits in base-2: t = s * (1/sqrt(64)) * log2(e)
  const float sc2 = 0.125f * 1.4426950408889634f;

  for (int kt = 0; kt < MK; kt += BN) {
    __syncthreads();  // previous iteration done reading sK/sVT

    // ---- Stage K (row-major) and V (transposed) tiles as f16 into LDS ----
    for (int i = tid; i < (BN * DH) / 4; i += 128) {
      const int key = (i * 4) >> 6;   // / DH
      const int d0  = (i * 4) & 63;
      v4f kv = *(const v4f*)&Kg[kt * DH + i * 4];
      v4f vv = *(const v4f*)&Vg[kt * DH + i * 4];
      v4h kh;
#pragma unroll
      for (int j = 0; j < 4; ++j) kh[j] = (_Float16)kv[j];
      *(v4h*)&sK[i * 4] = kh;
#pragma unroll
      for (int j = 0; j < 4; ++j) sVT[(d0 + j) * BN + key] = (_Float16)vv[j];
    }

    // ---- Prefetch next tile (global_prefetch_b8): one 128B line per thread ----
    if (kt + BN < MK) {
      __builtin_prefetch(&Kg[(kt + BN) * DH + tid * 32], 0, 0);
      __builtin_prefetch(&Vg[(kt + BN) * DH + tid * 32], 0, 0);
    }
    __syncthreads();

    // ---- S = Q * K^T (16 x 64): 4 key sub-tiles x 2 D-chunks ----
    v8f s[4];
#pragma unroll
    for (int n = 0; n < 4; ++n) {
      v8f acc = {};
#pragma unroll
      for (int c = 0; c < 2; ++c) {
        // B fragment (32x16): K = e + 16*hf + 32c over D, N = ln
        v16h b = *(const v16h*)&sK[(n * 16 + ln) * DH + hf * 16 + c * 32];
        acc = __builtin_amdgcn_wmma_f32_16x16x32_f16(false, qa[c], false, b,
                                                     (short)0, acc, false, false);
      }
      s[n] = acc;
    }

    // ---- Online softmax in exp2 domain (rows M = r + 8*hf; DPP reductions) ----
    float mnew[8], alpha[8], rs[8];
#pragma unroll
    for (int r = 0; r < 8; ++r) {
      float mx = fmaxf(fmaxf(s[0][r], s[1][r]), fmaxf(s[2][r], s[3][r]));
      mx = redmax16(mx * sc2);
      mnew[r]  = fmaxf(mrow[r], mx);
      alpha[r] = __builtin_amdgcn_exp2f(mrow[r] - mnew[r]);  // first iter: exp2(-inf)=0
      mrow[r]  = mnew[r];
      rs[r]    = 0.0f;
    }
#pragma unroll
    for (int n = 0; n < 4; ++n) {
#pragma unroll
      for (int r = 0; r < 8; ++r) {
        float p = __builtin_amdgcn_exp2f(__builtin_fmaf(s[n][r], sc2, -mnew[r]));
        s[n][r] = p;
        rs[r] += p;
      }
    }
#pragma unroll
    for (int r = 0; r < 8; ++r) lrow[r] = lrow[r] * alpha[r] + redsum16(rs[r]);
#pragma unroll
    for (int n = 0; n < 4; ++n) {
#pragma unroll
      for (int r = 0; r < 8; ++r) o[n][r] *= alpha[r];
    }

    // ---- Spill P (C-layout) into per-wave LDS as f16, row-major [qrow][key] ----
#pragma unroll
    for (int n = 0; n < 4; ++n) {
#pragma unroll
      for (int r = 0; r < 8; ++r) {
        sP[wave][(r + 8 * hf) * BN + n * 16 + ln] = (_Float16)s[n][r];
      }
    }
    // same-wave DS ordering protects the read-back below

    // ---- O += P * V : 2 key-chunks (K=32) x 4 DV sub-tiles ----
#pragma unroll
    for (int c = 0; c < 2; ++c) {
      // P A-fragment: two contiguous 8-half runs per lane
      const _Float16* pp = &sP[wave][ln * BN + 32 * c + 16 * hf];
      v8h plo = *(const v8h*)pp;
      v8h phi = *(const v8h*)(pp + 8);
      v16h pa;
#pragma unroll
      for (int j = 0; j < 8; ++j) { pa[j] = plo[j]; pa[8 + j] = phi[j]; }
#pragma unroll
      for (int n = 0; n < 4; ++n) {
        // B fragment from transposed V: contiguous over K
        v16h bv = *(const v16h*)&sVT[(n * 16 + ln) * BN + 16 * hf + 32 * c];
        o[n] = __builtin_amdgcn_wmma_f32_16x16x32_f16(false, pa, false, bv,
                                                      (short)0, o[n], false, false);
      }
    }
  }

  // ---- Epilogue: normalize with fast reciprocal (v_rcp_f32) and store fp32 ----
  float rinv[8];
#pragma unroll
  for (int r = 0; r < 8; ++r) rinv[r] = __builtin_amdgcn_rcpf(lrow[r]);
#pragma unroll
  for (int n = 0; n < 4; ++n) {
#pragma unroll
    for (int r = 0; r < 8; ++r) {
      Og[(qRow0 + r + 8 * hf) * DH + n * 16 + ln] = o[n][r] * rinv[r];
    }
  }
}

extern "C" void kernel_launch(void* const* d_in, const int* in_sizes, int n_in,
                              void* d_out, int out_size, void* d_ws, size_t ws_size,
                              hipStream_t stream) {
  const float* Q = (const float*)d_in[0];
  const float* K = (const float*)d_in[1];
  const float* V = (const float*)d_in[2];
  float* Out = (float*)d_out;
  (void)in_sizes; (void)n_in; (void)out_size; (void)d_ws; (void)ws_size;

  dim3 grid(NQ / BM);   // 128 workgroups
  dim3 block(128);      // 4 waves (wave32)
  fa_wmma_kernel<<<grid, block, 0, stream>>>(Q, K, V, Out);
}
